// TimeSeriesMamba_40896678592645
// MI455X (gfx1250) — matrix operand order repaired and checked by reading, not compiled, not run on hardware
//
#include <hip/hip_runtime.h>
#include <hip/hip_bf16.h>
#include <cstdint>

// ---------------- model dims ----------------
#define BATCH 8
#define SEQ   2048
#define DIN   4
#define DM    512
#define DSN   16          // state size (DS)
#define DCN   4           // conv taps (DC)
#define DI    1024        // EXP*DM
#define DTR   32
#define HID   1408
#define NLAY  4
#define TOK   (BATCH*SEQ) // 16384 tokens

typedef __attribute__((ext_vector_type(8)))  __bf16 v8bf;
typedef __attribute__((ext_vector_type(16))) __bf16 v16bf;
typedef __attribute__((ext_vector_type(8)))  float  v8f;
typedef int v4i_ __attribute__((vector_size(16)));

// ---- CDNA5 async global->LDS path (probe via __has_builtin; safe fallback) ----
#if defined(__has_builtin)
#  if __has_builtin(__builtin_amdgcn_global_load_async_to_lds_b128) && \
      __has_builtin(__builtin_amdgcn_s_wait_asynccnt)
#    define HAVE_ASYNC_LDS 1
#  endif
#endif
#ifndef HAVE_ASYNC_LDS
#  define HAVE_ASYNC_LDS 0
#endif

__device__ __forceinline__ void stage_b128(const __bf16* g, __bf16* l) {
#if HAVE_ASYNC_LDS
    __builtin_amdgcn_global_load_async_to_lds_b128(
        (__attribute__((address_space(1))) v4i_*)(uintptr_t)g,
        (__attribute__((address_space(3))) v4i_*)(uint32_t)(uintptr_t)l,
        0, 0);
#else
    *(uint4*)l = *(const uint4*)g;   // global_load_b128 + ds_store_b128
#endif
}
__device__ __forceinline__ void wait_async_le(int n) {
#if HAVE_ASYNC_LDS
    if (n == 0) __builtin_amdgcn_s_wait_asynccnt(0);
    else        __builtin_amdgcn_s_wait_asynccnt(1);
#else
    (void)n;
#endif
}

__device__ __forceinline__ float sigmoidf_(float v) { return 1.f / (1.f + expf(-v)); }

// ---------------- f32 -> bf16 convert ----------------
__global__ void f32_to_bf16_kernel(const float* __restrict__ in, __bf16* __restrict__ out, long n) {
    long i = (long)blockIdx.x * blockDim.x + threadIdx.x;
    long stride = (long)gridDim.x * blockDim.x;
    for (; i < n; i += stride) out[i] = (__bf16)in[i];
}

// ---------------- embed ----------------
__global__ void embed_kernel(const float* __restrict__ x, const float* __restrict__ ew,
                             const float* __restrict__ eb, float* __restrict__ h) {
    size_t i = (size_t)blockIdx.x * 256 + threadIdx.x;   // TOK*DM threads
    size_t token = i >> 9;
    int m = (int)(i & (DM - 1));
    const float* xp = x + token * DIN;
    const float* wp = ew + (size_t)m * DIN;
    float s = eb[m];
#pragma unroll
    for (int k = 0; k < DIN; k++) s += xp[k] * wp[k];
    h[i] = s;
}

// ---------------- residual add + LayerNorm -> bf16 (one wave per token) ----------------
__global__ void __launch_bounds__(256)
add_ln_kernel(const float* __restrict__ a, float* __restrict__ res,
              const float* __restrict__ w, const float* __restrict__ bias,
              __bf16* __restrict__ out, int add) {
    int wave = threadIdx.x >> 5;
    int lane = threadIdx.x & 31;
    size_t token = (size_t)blockIdx.x * 8 + wave;
    size_t base = token * DM;
    float v[16];
    float s = 0.f;
#pragma unroll
    for (int j = 0; j < 16; j++) {
        size_t i = base + lane + j * 32;
        float r = a[i];
        if (add) r += res[i];
        res[i] = r;
        v[j] = r;
        s += r;
    }
    for (int off = 16; off; off >>= 1) s += __shfl_xor(s, off, 32);
    float mu = s * (1.f / DM);
    float ss = 0.f;
#pragma unroll
    for (int j = 0; j < 16; j++) { float d = v[j] - mu; ss += d * d; }
    for (int off = 16; off; off >>= 1) ss += __shfl_xor(ss, off, 32);
    float rinv = rsqrtf(ss * (1.f / DM) + 1e-5f);
#pragma unroll
    for (int j = 0; j < 16; j++) {
        int c = lane + j * 32;
        out[base + c] = (__bf16)((v[j] - mu) * rinv * w[c] + bias[c]);
    }
}

// ---------------- bf16 WMMA GEMM (NT), LDS-staged B, double-buffered ----------------
// C[M,N] (f32) = A[M,K](bf16,row-major,lda) x W[N,K]^T (bf16,row-major,ldw)
// block: 256 thr = 8 waves (4M x 2N); wave tile 32x32 (2x2 WMMA); block tile 128x64.
// B tile (64 cols x 32 k = 4KB) ping-pong staged in LDS via async-to-LDS.
// grid (N/64, M/128). Requires K%32==0, N%64==0, M%128==0.
__global__ void __launch_bounds__(256)
gemm_bf16_nt(const __bf16* __restrict__ A, int lda,
             const __bf16* __restrict__ W, int ldw,
             float* __restrict__ C, int ldc, int K) {
    __shared__ __bf16 Bsm[2][64][32];

    const int tid  = threadIdx.x;
    const int lane = tid & 31;
    const int wave = tid >> 5;
    const int wm = wave & 3;      // 0..3 (M)
    const int wn = wave >> 2;     // 0..1 (N)
    const int row0 = blockIdx.y * 128 + wm * 32;
    const int colg = blockIdx.x * 64;
    const int colL = wn * 32;

    // B staging: thread tid copies 16B of W row (colg+sn), k chunk sc
    const int sn = tid >> 2;            // 0..63
    const int sc = (tid & 3) * 8;       // 0,8,16,24
    const __bf16* wsrc = W + (size_t)(colg + sn) * ldw + sc;

    // A fragment addressing (16x32 bf16 layout)
    const int am  = lane & 15;
    const int akb = (lane < 16) ? 0 : 8;
    const __bf16* ap0 = A + (size_t)(row0 + am) * lda + akb;
    const __bf16* ap1 = ap0 + (size_t)16 * lda;

    // B fragment addressing from LDS (32x16 bf16 layout)
    const int bn  = lane & 15;
    const int bkb = (lane < 16) ? 0 : 8;

    const v8f vzero = {0.f,0.f,0.f,0.f,0.f,0.f,0.f,0.f};
    v8f acc00 = vzero, acc01 = vzero, acc10 = vzero, acc11 = vzero;

    const int nk = K / 32;

    // ---- prologue: stage B tile 0, load A frags for k=0 ----
    stage_b128(wsrc, &Bsm[0][sn][sc]);
    v8bf ca0 = *(const v8bf*)(ap0);
    v8bf ca1 = *(const v8bf*)(ap0 + 16);
    v8bf ca2 = *(const v8bf*)(ap1);
    v8bf ca3 = *(const v8bf*)(ap1 + 16);

    for (int ks = 0; ks < nk; ks++) {
        const int kn = (ks + 1) * 32;
        v8bf na0, na1, na2, na3;
        if (ks + 1 < nk) {
            // issue next B tile (other LDS buffer) + next A frags before consuming current
            stage_b128(wsrc + kn, &Bsm[(ks + 1) & 1][sn][sc]);
            na0 = *(const v8bf*)(ap0 + kn);
            na1 = *(const v8bf*)(ap0 + kn + 16);
            na2 = *(const v8bf*)(ap1 + kn);
            na3 = *(const v8bf*)(ap1 + kn + 16);
            __builtin_prefetch(ap0 + kn + 32, 0, 3);   // global_prefetch_b8
            __builtin_prefetch(ap1 + kn + 32, 0, 3);
            wait_async_le(1);    // my stage-ks copy complete (next one still in flight)
        } else {
            wait_async_le(0);
        }
        __syncthreads();         // everyone's stage-ks B tile visible in LDS

        const int buf = ks & 1;
        const __bf16* b0p = &Bsm[buf][colL + bn][bkb];
        const __bf16* b1p = &Bsm[buf][colL + 16 + bn][bkb];
        v8bf b0lo = *(const v8bf*)(b0p);
        v8bf b0hi = *(const v8bf*)(b0p + 16);
        v8bf b1lo = *(const v8bf*)(b1p);
        v8bf b1hi = *(const v8bf*)(b1p + 16);

        v16bf fa0 = __builtin_shufflevector(ca0, ca1, 0,1,2,3,4,5,6,7,8,9,10,11,12,13,14,15);
        v16bf fa1 = __builtin_shufflevector(ca2, ca3, 0,1,2,3,4,5,6,7,8,9,10,11,12,13,14,15);
        v16bf fb0 = __builtin_shufflevector(b0lo, b0hi, 0,1,2,3,4,5,6,7,8,9,10,11,12,13,14,15);
        v16bf fb1 = __builtin_shufflevector(b1lo, b1hi, 0,1,2,3,4,5,6,7,8,9,10,11,12,13,14,15);

        acc00 = __builtin_amdgcn_wmma_f32_16x16x32_bf16(false, fa0, false, fb0, (short)0, acc00, false, false);
        acc01 = __builtin_amdgcn_wmma_f32_16x16x32_bf16(false, fa0, false, fb1, (short)0, acc01, false, false);
        acc10 = __builtin_amdgcn_wmma_f32_16x16x32_bf16(false, fa1, false, fb0, (short)0, acc10, false, false);
        acc11 = __builtin_amdgcn_wmma_f32_16x16x32_bf16(false, fa1, false, fb1, (short)0, acc11, false, false);

        __syncthreads();         // all waves done reading buf before it is re-staged
        ca0 = na0; ca1 = na1; ca2 = na2; ca3 = na3;
    }

    // D layout: lane<16 -> rows 0..7, N=lane; lane>=16 -> rows 8..15, N=lane-16
    const int n = lane & 15;
    const int mbase = (lane >> 4) << 3;
#pragma unroll
    for (int r = 0; r < 8; r++) {
        size_t r0 = (size_t)(row0 + mbase + r) * ldc + colg + colL + n;
        size_t r1 = (size_t)(row0 + 16 + mbase + r) * ldc + colg + colL + n;
        C[r0]      = acc00[r];
        C[r0 + 16] = acc01[r];
        C[r1]      = acc10[r];
        C[r1 + 16] = acc11[r];
    }
}

// ---------------- depthwise causal conv (DC=4) + SiLU ----------------
__global__ void conv_silu_kernel(const float* __restrict__ xz, const float* __restrict__ cw,
                                 const float* __restrict__ cb, float* __restrict__ actf,
                                 __bf16* __restrict__ actbf) {
    size_t i = (size_t)blockIdx.x * 256 + threadIdx.x;   // TOK*DI threads
    size_t token = i / DI;
    int e = (int)(i - token * DI);
    int b = (int)(token / SEQ);
    int t = (int)(token - (size_t)b * SEQ);
    float s = cb[e];
#pragma unroll
    for (int k = 0; k < DCN; k++) {
        int tt = t + k - (DCN - 1);
        if (tt >= 0)
            s += cw[e * DCN + k] * xz[((size_t)b * SEQ + tt) * (2 * DI) + e];
    }
    float a = s * sigmoidf_(s);
    actf[i] = a;
    actbf[i] = (__bf16)a;
}

// ---------------- selective scan (sequential in L) ----------------
__global__ void __launch_bounds__(256)
scan_kernel(const float* __restrict__ dtraw, const float* __restrict__ dtb,
            const float* __restrict__ xdbl, const float* __restrict__ A_log,
            const float* __restrict__ Dp, const float* __restrict__ actf,
            const float* __restrict__ xz, __bf16* __restrict__ ybf) {
    int d = blockIdx.x * 256 + threadIdx.x;
    int b = blockIdx.y;
    float negA[DSN];
#pragma unroll
    for (int s = 0; s < DSN; s++) negA[s] = -expf(A_log[(size_t)d * DSN + s]);
    float h[DSN];
#pragma unroll
    for (int s = 0; s < DSN; s++) h[s] = 0.f;
    float dtbias = dtb[d];
    float Dd = Dp[d];
    __shared__ float Bs[DSN];
    __shared__ float Cs[DSN];

    for (int t = 0; t < SEQ; t++) {
        size_t row = (size_t)b * SEQ + t;
        if (threadIdx.x < 2 * DSN) {
            float v = xdbl[row * (DTR + 2 * DSN) + DTR + threadIdx.x];
            if (threadIdx.x < DSN) Bs[threadIdx.x] = v;
            else Cs[threadIdx.x - DSN] = v;
        }
        __syncthreads();
        float dr = dtraw[row * DI + d] + dtbias;
        float dt = (dr > 20.f) ? dr : log1pf(expf(dr));   // softplus
        float u = actf[row * DI + d];
        float y = 0.f;
#pragma unroll
        for (int s = 0; s < DSN; s++) {
            float dA = expf(dt * negA[s]);
            h[s] = dA * h[s] + dt * Bs[s] * u;
            y += h[s] * Cs[s];
        }
        y += Dd * u;
        float z = xz[row * (2 * DI) + DI + d];
        y *= z * sigmoidf_(z);
        ybf[row * DI + d] = (__bf16)y;
        __syncthreads();
    }
}

// ---------------- SwiGLU ----------------
__global__ void swiglu_kernel(const float* __restrict__ y1, __bf16* __restrict__ g1) {
    size_t i = (size_t)blockIdx.x * 256 + threadIdx.x;   // TOK*HID threads
    size_t row = i / HID;
    int hc = (int)(i - row * HID);
    float a = y1[row * (2 * HID) + hc];
    float g = y1[row * (2 * HID) + HID + hc];
    g1[i] = (__bf16)(a * g * sigmoidf_(g));
}

// ---------------- final LN + prediction head ----------------
__global__ void __launch_bounds__(256)
final_kernel(const float* __restrict__ h, const float* __restrict__ fw,
             const float* __restrict__ fb, const float* __restrict__ pw,
             const float* __restrict__ pb, float* __restrict__ out) {
    int wave = threadIdx.x >> 5;
    int lane = threadIdx.x & 31;
    size_t base = ((size_t)wave * SEQ + (SEQ - 1)) * DM;
    float v[16];
    float s = 0.f;
#pragma unroll
    for (int j = 0; j < 16; j++) { v[j] = h[base + lane + j * 32]; s += v[j]; }
    for (int off = 16; off; off >>= 1) s += __shfl_xor(s, off, 32);
    float mu = s * (1.f / DM);
    float ss = 0.f;
#pragma unroll
    for (int j = 0; j < 16; j++) { float d = v[j] - mu; ss += d * d; }
    for (int off = 16; off; off >>= 1) ss += __shfl_xor(ss, off, 32);
    float rinv = rsqrtf(ss * (1.f / DM) + 1e-5f);
    float acc = 0.f;
#pragma unroll
    for (int j = 0; j < 16; j++) {
        int c = lane + j * 32;
        float nz = (v[j] - mu) * rinv * fw[c] + fb[c];
        acc += nz * pw[c];
    }
    for (int off = 16; off; off >>= 1) acc += __shfl_xor(acc, off, 32);
    if (lane == 0) out[wave] = acc + pb[0];
}

// ---------------- host launch ----------------
extern "C" void kernel_launch(void* const* d_in, const int* in_sizes, int n_in,
                              void* d_out, int out_size, void* d_ws, size_t ws_size,
                              hipStream_t stream) {
    (void)in_sizes; (void)n_in; (void)out_size; (void)ws_size;
    const float* x_in       = (const float*)d_in[0];
    const float* embed_w    = (const float*)d_in[1];
    const float* embed_b    = (const float*)d_in[2];
    const float* in_proj_w  = (const float*)d_in[3];
    const float* conv_w     = (const float*)d_in[4];
    const float* conv_b     = (const float*)d_in[5];
    const float* x_proj_w   = (const float*)d_in[6];
    const float* dt_proj_w  = (const float*)d_in[7];
    const float* dt_proj_b  = (const float*)d_in[8];
    const float* A_log      = (const float*)d_in[9];
    const float* Dparam     = (const float*)d_in[10];
    const float* out_proj_w = (const float*)d_in[11];
    const float* norm1_w    = (const float*)d_in[12];
    const float* norm1_b    = (const float*)d_in[13];
    const float* norm2_w    = (const float*)d_in[14];
    const float* norm2_b    = (const float*)d_in[15];
    const float* fc1_w      = (const float*)d_in[16];
    const float* fc2_w      = (const float*)d_in[17];
    const float* fnw        = (const float*)d_in[18];
    const float* fnb        = (const float*)d_in[19];
    const float* pred_w     = (const float*)d_in[20];
    const float* pred_b     = (const float*)d_in[21];
    float* out = (float*)d_out;

    // -------- carve workspace (256B aligned) --------
    size_t off = 0;
    auto carve = [&](size_t bytes) -> char* {
        off = (off + 255) & ~(size_t)255;
        char* p = (char*)d_ws + off;
        off += bytes;
        return p;
    };
    const size_t T = TOK;
    __bf16* w_in  = (__bf16*)carve((size_t)NLAY * 2 * DI * DM * 2);
    __bf16* w_x   = (__bf16*)carve((size_t)NLAY * (DTR + 2 * DSN) * DI * 2);
    __bf16* w_dt  = (__bf16*)carve((size_t)NLAY * DI * DTR * 2);
    __bf16* w_o   = (__bf16*)carve((size_t)NLAY * DM * DI * 2);
    __bf16* w_f1  = (__bf16*)carve((size_t)NLAY * 2 * HID * DM * 2);
    __bf16* w_f2  = (__bf16*)carve((size_t)NLAY * DM * HID * 2);
    float*  res   = (float*)carve(T * DM * 4);
    float*  hbuf  = (float*)carve(T * DM * 4);
    __bf16* hsbf  = (__bf16*)carve(T * DM * 2);
    float*  big   = (float*)carve(T * 2 * HID * 4);     // xz (T,2048) / y1 (T,2816)
    float*  actf  = (float*)carve(T * DI * 4);
    __bf16* shbf  = (__bf16*)carve(T * HID * 2);        // act_bf (T,1024) / g1 (T,1408)
    float*  xdbl  = (float*)carve(T * (DTR + 2 * DSN) * 4);
    __bf16* xdblb = (__bf16*)carve(T * (DTR + 2 * DSN) * 2);
    float*  dtmo  = (float*)carve(T * DI * 4);          // dtraw (T,1024) / mamba_out (T,512)
    __bf16* ybf   = (__bf16*)carve(T * DI * 2);

    auto cvt = [&](const float* src, __bf16* dst, long n) {
        int blocks = (int)((n + 1023) / 1024);
        if (blocks > 65535) blocks = 65535;
        f32_to_bf16_kernel<<<blocks, 256, 0, stream>>>(src, dst, n);
    };
    cvt(in_proj_w,  w_in, (long)NLAY * 2 * DI * DM);
    cvt(x_proj_w,   w_x,  (long)NLAY * (DTR + 2 * DSN) * DI);
    cvt(dt_proj_w,  w_dt, (long)NLAY * DI * DTR);
    cvt(out_proj_w, w_o,  (long)NLAY * DM * DI);
    cvt(fc1_w,      w_f1, (long)NLAY * 2 * HID * DM);
    cvt(fc2_w,      w_f2, (long)NLAY * DM * HID);

    auto gemm = [&](const __bf16* A, int lda, const __bf16* W, int ldw, float* C, int N, int K) {
        dim3 g(N / 64, (int)(T / 128));
        gemm_bf16_nt<<<g, 256, 0, stream>>>(A, lda, W, ldw, C, N, K);
    };

    // embed
    embed_kernel<<<(int)(T * DM / 256), 256, 0, stream>>>(x_in, embed_w, embed_b, hbuf);

    for (int l = 0; l < NLAY; l++) {
        // res (+)= h ; hs = LN1(res)
        add_ln_kernel<<<(int)(T / 8), 256, 0, stream>>>(hbuf, res, norm1_w + (size_t)l * DM,
                                                        norm1_b + (size_t)l * DM, hsbf, l > 0);
        // in_proj: (T,512)x(2048,512)^T -> xz
        gemm(hsbf, DM, w_in + (size_t)l * 2 * DI * DM, DM, big, 2 * DI, DM);
        // conv + silu
        conv_silu_kernel<<<(int)(T * DI / 256), 256, 0, stream>>>(
            big, conv_w + (size_t)l * DI * DCN, conv_b + (size_t)l * DI, actf, shbf);
        // x_proj: (T,1024)x(64,1024)^T -> xdbl
        gemm(shbf, DI, w_x + (size_t)l * (DTR + 2 * DSN) * DI, DI, xdbl, DTR + 2 * DSN, DI);
        cvt(xdbl, xdblb, (long)T * (DTR + 2 * DSN));
        // dt_proj: (T,32)x(1024,32)^T -> dtraw
        gemm(xdblb, DTR + 2 * DSN, w_dt + (size_t)l * DI * DTR, DTR, dtmo, DI, DTR);
        // selective scan -> y (bf16)
        dim3 sg(DI / 256, BATCH);
        scan_kernel<<<sg, 256, 0, stream>>>(dtmo, dt_proj_b + (size_t)l * DI, xdbl,
                                            A_log + (size_t)l * DI * DSN,
                                            Dparam + (size_t)l * DI, actf, big, ybf);
        // out_proj: (T,1024)x(512,1024)^T -> mamba_out (aliases dtraw)
        gemm(ybf, DI, w_o + (size_t)l * DM * DI, DI, dtmo, DM, DI);
        // res += mamba_out ; hs = LN2(res)
        add_ln_kernel<<<(int)(T / 8), 256, 0, stream>>>(dtmo, res, norm2_w + (size_t)l * DM,
                                                        norm2_b + (size_t)l * DM, hsbf, 1);
        // fc1: (T,512)x(2816,512)^T -> y1 (aliases xz)
        gemm(hsbf, DM, w_f1 + (size_t)l * 2 * HID * DM, DM, big, 2 * HID, DM);
        // swiglu -> g1 bf16
        swiglu_kernel<<<(int)(T * HID / 256), 256, 0, stream>>>(big, shbf);
        // fc2: (T,1408)x(512,1408)^T -> h
        gemm(shbf, HID, w_f2 + (size_t)l * DM * HID, HID, hbuf, DM, HID);
    }

    final_kernel<<<1, 256, 0, stream>>>(hbuf, fnw, fnb, pred_w, pred_b, out);
}
